// S4_38680475468251
// MI455X (gfx1250) — matrix operand order repaired and checked by reading, not compiled, not run on hardware
//
#include <hip/hip_runtime.h>
#include <math.h>
#include <stdint.h>

// Problem constants (B, C, H, L) = (8, 1, 512, 8192), FFT length 16384.
#define BATCH 8
#define HCH   512
#define LSEQ  8192
#define NFFT  16384
#define FFT_T 512   // threads per FFT workgroup

typedef __attribute__((ext_vector_type(16))) __bf16 v16bf;
typedef __attribute__((ext_vector_type(8)))  __bf16 v8bf;
typedef __attribute__((ext_vector_type(8)))  float  v8f;
typedef __attribute__((ext_vector_type(4))) unsigned int u32x4;
typedef __attribute__((ext_vector_type(8))) int i32x8;
typedef __attribute__((ext_vector_type(4))) int i32x4;

struct c32 { float x, y; };

__device__ __forceinline__ c32 cadd(c32 a, c32 b) { return {a.x + b.x, a.y + b.y}; }
__device__ __forceinline__ c32 csub(c32 a, c32 b) { return {a.x - b.x, a.y - b.y}; }
__device__ __forceinline__ c32 cmul(c32 a, c32 b) {
  return {a.x * b.x - a.y * b.y, a.x * b.y + a.y * b.x};
}
// a * conj(b)
__device__ __forceinline__ c32 cmulc(c32 a, c32 b) {
  return {a.x * b.x + a.y * b.y, a.y * b.x - a.x * b.y};
}

__device__ __forceinline__ unsigned short f2bf(float x) {
  unsigned int u = __float_as_uint(x);
  u = (u + 0x7FFFu + ((u >> 16) & 1u)) >> 16;
  return (unsigned short)u;
}

// ---------------------------------------------------------------------------
// Kernel 0: twiddle table tw[j] = exp(-2*pi*i*j/NFFT), j < NFFT/2
// ---------------------------------------------------------------------------
__global__ void s4_tw_init(c32* __restrict__ tw) {
  int i = blockIdx.x * blockDim.x + threadIdx.x;
  if (i < NFFT / 2) {
    float ang = -6.283185307179586f * (float)i / (float)NFFT;
    float s, c;
    sincosf(ang, &s, &c);
    tw[i] = {c, s};
  }
}

// ---------------------------------------------------------------------------
// Kernel 1: W (fp32, [o][h]) -> bf16
// ---------------------------------------------------------------------------
__global__ void s4_wprep(const float* __restrict__ W, unsigned short* __restrict__ Wb) {
  int i = blockIdx.x * blockDim.x + threadIdx.x;
  if (i < HCH * HCH) Wb[i] = f2bf(W[i]);
}

// ---------------------------------------------------------------------------
// In-LDS 16384-pt FFT, 512 threads.
// Forward: radix-2 DIF, natural in -> bit-reversed out; two stages fused per
// LDS pass (7 barriers). Inverse: radix-2 DIT, conjugate twiddles, natural out.
// Pointwise products in bit-reversed order are consistent => no reversal pass.
// ---------------------------------------------------------------------------
__device__ void fft_fwd(c32* s, const c32* __restrict__ tw, int tid) {
  for (int M1 = NFFT / 2; M1 >= 2; M1 >>= 2) {
    int M2 = M1 >> 1;
    int strA = (NFFT / 2) / M1;
    for (int q = tid; q < NFFT / 4; q += FFT_T) {
      int a = q / M2, j = q - a * M2;
      int i0 = a * (2 * M1) + j;
      c32 p0 = s[i0], p1 = s[i0 + M2], p2 = s[i0 + M1], p3 = s[i0 + M1 + M2];
      c32 wA0 = tw[j * strA];
      c32 wA1 = tw[(j + M2) * strA];
      c32 wB  = tw[j * (2 * strA)];
      c32 u0 = cadd(p0, p2), u2 = cmul(csub(p0, p2), wA0);
      c32 u1 = cadd(p1, p3), u3 = cmul(csub(p1, p3), wA1);
      s[i0]           = cadd(u0, u1);
      s[i0 + M2]      = cmul(csub(u0, u1), wB);
      s[i0 + M1]      = cadd(u2, u3);
      s[i0 + M1 + M2] = cmul(csub(u2, u3), wB);
    }
    __syncthreads();
  }
}

__device__ void fft_inv(c32* s, const c32* __restrict__ tw, int tid) {
  for (int M2 = 1; M2 <= NFFT / 4; M2 <<= 2) {
    int M1 = M2 << 1;
    int strA = (NFFT / 2) / M1;
    for (int q = tid; q < NFFT / 4; q += FFT_T) {
      int a = q / M2, j = q - a * M2;
      int i0 = a * (2 * M1) + j;
      c32 p0 = s[i0], p1 = s[i0 + M2], p2 = s[i0 + M1], p3 = s[i0 + M1 + M2];
      c32 wA0 = tw[j * strA];
      c32 wA1 = tw[(j + M2) * strA];
      c32 wB  = tw[j * (2 * strA)];
      c32 t;
      t = cmulc(p1, wB); c32 u0 = cadd(p0, t), u1 = csub(p0, t);
      t = cmulc(p3, wB); c32 u2 = cadd(p2, t), u3 = csub(p2, t);
      t = cmulc(u2, wA0); s[i0]      = cadd(u0, t); s[i0 + M1]      = csub(u0, t);
      t = cmulc(u3, wA1); s[i0 + M2] = cadd(u1, t); s[i0 + M1 + M2] = csub(u1, t);
    }
    __syncthreads();
  }
}

// ---------------------------------------------------------------------------
// Kernel 2: spectrum of conv kernel k, scaled by 1/NFFT (folds irfft scaling)
// ---------------------------------------------------------------------------
__global__ void s4_kfft(const float* __restrict__ k, c32* __restrict__ Kf,
                        const c32* __restrict__ tw) {
  extern __shared__ c32 s[];
  int h = blockIdx.x, tid = threadIdx.x;
  const float* kr = k + (size_t)h * LSEQ;
  for (int i = tid; i < NFFT; i += FFT_T) s[i] = {(i < LSEQ) ? kr[i] : 0.0f, 0.0f};
  __syncthreads();
  fft_fwd(s, tw, tid);
  const float sc = 1.0f / (float)NFFT;
  c32* outp = Kf + (size_t)h * NFFT;
  for (int i = tid; i < NFFT; i += FFT_T) {
    c32 v = s[i];
    outp[i] = {v.x * sc, v.y * sc};
  }
}

// ---------------------------------------------------------------------------
// Kernel 3: per-(b,h) FFT conv + D skip + exact GELU -> bf16 activations g
// ---------------------------------------------------------------------------
__global__ void s4_conv(const float* __restrict__ u, const c32* __restrict__ Kf,
                        const float* __restrict__ D, unsigned short* __restrict__ g,
                        const c32* __restrict__ tw) {
  extern __shared__ c32 s[];
  int bh = blockIdx.x;
  int h = bh & (HCH - 1);
  int tid = threadIdx.x;
  const float* ur = u + (size_t)bh * LSEQ;
  const c32* kf = Kf + (size_t)h * NFFT;
  for (int i = tid; i < NFFT; i += FFT_T) s[i] = {(i < LSEQ) ? ur[i] : 0.0f, 0.0f};
  // pull this block's Kf row (128 KB) toward the caches while the forward FFT
  // runs: one 128B line per thread per step => global_prefetch_b8
  for (int i = tid * 16; i < NFFT; i += FFT_T * 16) __builtin_prefetch(&kf[i], 0, 1);
  __syncthreads();
  fft_fwd(s, tw, tid);
  for (int i = tid; i < NFFT; i += FFT_T) s[i] = cmul(s[i], kf[i]);
  __syncthreads();
  fft_inv(s, tw, tid);
  float Dh = D[h];  // C == 1
  unsigned short* gr = g + (size_t)bh * LSEQ;
  for (int i = tid; i < LSEQ; i += FFT_T) {
    float x = s[i].x + Dh * ur[i];              // irfft already scaled via Kf
    float gv = 0.5f * x * (1.0f + erff(x * 0.70710678118654752f));
    gr[i] = f2bf(gv);
  }
}

// ---------------------------------------------------------------------------
// Kernel 4: g (b,h,l) -> gt (b,l,h), bf16, 32x32 LDS tiles
// ---------------------------------------------------------------------------
__global__ void s4_transpose(const unsigned short* __restrict__ g,
                             unsigned short* __restrict__ gt) {
  __shared__ unsigned short t[32][33];
  int b = blockIdx.z;
  int l0 = blockIdx.x * 32, h0 = blockIdx.y * 32;
  int tx = threadIdx.x, ty = threadIdx.y;
  const unsigned short* gb = g + (size_t)b * HCH * LSEQ;
  unsigned short* gtb = gt + (size_t)b * LSEQ * HCH;
#pragma unroll
  for (int r = 0; r < 4; ++r) {
    int hh = h0 + ty + r * 8;
    t[ty + r * 8][tx] = gb[(size_t)hh * LSEQ + l0 + tx];
  }
  __syncthreads();
#pragma unroll
  for (int r = 0; r < 4; ++r) {
    int ll = l0 + ty + r * 8;
    gtb[(size_t)ll * HCH + h0 + tx] = t[tx][ty + r * 8];
  }
}

// ---------------------------------------------------------------------------
// Tensor Data Mover: 2-D tile (tile_d0 x tile_d1 elements, 2-byte elements)
// global -> LDS, with LDS padding 16 DWORDs data + 4 DWORDs pad so a 64B row
// lands at an 80B stride (= LDA/LDB of 40 halves, bank-conflict-free frags).
// D# layout per CDNA5 ISA 8.3/8.4 (group0 128b, group1 256b; groups 2/3 zero).
// This toolchain exposes the 6-arg builtin (extra int32x8 group, zero-filled).
// ---------------------------------------------------------------------------
__device__ __forceinline__ void tdm_load_2d(unsigned lds_off, const void* gptr,
                                            unsigned tile_d0, unsigned tile_d1,
                                            unsigned tensor_d0, unsigned tensor_d1,
                                            unsigned stride0_elems) {
  unsigned long long ga = (unsigned long long)(uintptr_t)gptr;
  u32x4 g0;
  g0[0] = 1u;                                    // count=1, user descriptor
  g0[1] = lds_off;                               // lds_addr (bytes)
  g0[2] = (unsigned)(ga & 0xFFFFFFFFu);          // global_addr[31:0]
  g0[3] = (unsigned)((ga >> 32) & 0x01FFFFFFu)   // global_addr[56:32]
        | (2u << 30);                            // type = 2 ("image")
  i32x8 g1;
  g1[0] = (int)((1u << 16)                       // data_size = 1 -> 2 bytes
              | (1u << 20)                       // pad_enable
              | (3u << 22)                       // pad_interval: 16 DWORDs
              | (3u << 25));                     // pad_amount: 4 DWORDs
  g1[1] = (int)((tensor_d0 & 0xFFFFu) << 16);                                  // [63:48] tensor_dim0 lo
  g1[2] = (int)(((tensor_d0 >> 16) & 0xFFFFu) | ((tensor_d1 & 0xFFFFu) << 16)); // dim0 hi | dim1 lo
  g1[3] = (int)(((tensor_d1 >> 16) & 0xFFFFu) | ((tile_d0 & 0xFFFFu) << 16));   // dim1 hi | tile_dim0
  g1[4] = (int)(tile_d1 & 0xFFFFu);              // tile_dim1 (tile_dim2 = 0)
  g1[5] = (int)stride0_elems;                    // tensor_dim0_stride[31:0]
  g1[6] = 0;                                     // stride0 hi | dim1_stride lo
  g1[7] = 0;
  i32x4 z4 = {0, 0, 0, 0};
  i32x8 z8 = {0, 0, 0, 0, 0, 0, 0, 0};
  __builtin_amdgcn_tensor_load_to_lds(g0, g1, z4, z4, z8, 0);
}

// ---------------------------------------------------------------------------
// Kernel 5: out[b,o,l] = sum_h W[o,h] * g[b,h,l] + bias[o]  (bf16 WMMA GEMM)
// M=512(o), K=512(h), N=B*L=65536. Block 128x128, 8 waves (2x4), wave 64x32,
// 4x2 frags of v_wmma_f32_16x16x32_bf16. A/B tiles staged by TDM, double-
// buffered in LDS: DMA for step t+1 overlaps the WMMAs of step t.
// ---------------------------------------------------------------------------
#define BM 128
#define BN 128
#define BK 32
#define LDA 40  // halves per LDS row (64B data + 16B TDM pad = 80B)
#define LDB 40
#define KSTEPS (HCH / BK)

__global__ __launch_bounds__(256) void s4_gemm(const __bf16* __restrict__ Wb,
                                               const __bf16* __restrict__ gt,
                                               const float* __restrict__ bias,
                                               float* __restrict__ outp) {
  __shared__ __bf16 As[2][BM * LDA];
  __shared__ __bf16 Bs[2][BN * LDB];
  int tid = threadIdx.x;
  int m0 = (blockIdx.x & 3) * BM;   // 512/128 = 4 M-tiles
  int nblk = blockIdx.x >> 2;       // 65536/128 = 512 N-tiles
  int b = nblk >> 6;                // 8192/128 = 64 N-tiles per batch
  int l0 = (nblk & 63) * BN;

  const __bf16* Abase = Wb + (size_t)m0 * HCH;                 // [o][h]
  const __bf16* Bbase = gt + ((size_t)b * LSEQ + l0) * HCH;    // [l][h]
  int wave = tid >> 5, lane = tid & 31;
  int wm = wave >> 2, wn = wave & 3;
  int lr = lane & 15, gsel = lane >> 4;

  // wave-uniform issuer guard (scalar branch: TDM ignores EXEC)
  bool issuer = ((__builtin_amdgcn_readfirstlane(tid) >> 5) == 0);
  unsigned aoff[2], boff[2];
  aoff[0] = (unsigned)(uintptr_t)&As[0][0];
  aoff[1] = (unsigned)(uintptr_t)&As[1][0];
  boff[0] = (unsigned)(uintptr_t)&Bs[0][0];
  boff[1] = (unsigned)(uintptr_t)&Bs[1][0];

  v8f acc[4][2];
#pragma unroll
  for (int i = 0; i < 4; ++i)
#pragma unroll
    for (int j = 0; j < 2; ++j) acc[i][j] = (v8f){0.f, 0.f, 0.f, 0.f, 0.f, 0.f, 0.f, 0.f};

  if (issuer) {
    tdm_load_2d(aoff[0], Abase, BK, BM, HCH, HCH, HCH);
    tdm_load_2d(boff[0], Bbase, BK, BN, HCH, HCH, HCH);
  }

  for (int t = 0; t < KSTEPS; ++t) {
    int cur = t & 1, nxt = cur ^ 1;
    if (issuer) {
      if (t + 1 < KSTEPS) {
        tdm_load_2d(aoff[nxt], Abase + (t + 1) * BK, BK, BM, HCH, HCH, HCH);
        tdm_load_2d(boff[nxt], Bbase + (t + 1) * BK, BK, BN, HCH, HCH, HCH);
        __builtin_amdgcn_s_wait_tensorcnt(2);   // pair t done, pair t+1 in flight
      } else {
        __builtin_amdgcn_s_wait_tensorcnt(0);
      }
    }
    __syncthreads();

    const __bf16* Asc = &As[cur][0];
    const __bf16* Bsc = &Bs[cur][0];
    // A frags: 16x32, lane holds row M=lr; e<8 -> K=gsel*8+e, e>=8 -> K=16+gsel*8+(e-8)
    v16bf af[4];
#pragma unroll
    for (int i = 0; i < 4; ++i) {
      int m = wm * 64 + i * 16 + lr;
      v8bf lo = *(const v8bf*)&Asc[m * LDA + gsel * 8];
      v8bf hi = *(const v8bf*)&Asc[m * LDA + 16 + gsel * 8];
      af[i] = __builtin_shufflevector(lo, hi, 0, 1, 2, 3, 4, 5, 6, 7, 8, 9, 10, 11, 12, 13, 14, 15);
    }
    // B frags: 32x16, lane holds col N=lr; e<8 -> K=gsel*8+e, e>=8 -> K=16+gsel*8+(e-8)
    v16bf bfr[2];
#pragma unroll
    for (int j = 0; j < 2; ++j) {
      int n = wn * 32 + j * 16 + lr;
      v8bf lo = *(const v8bf*)&Bsc[n * LDB + gsel * 8];
      v8bf hi = *(const v8bf*)&Bsc[n * LDB + 16 + gsel * 8];
      bfr[j] = __builtin_shufflevector(lo, hi, 0, 1, 2, 3, 4, 5, 6, 7, 8, 9, 10, 11, 12, 13, 14, 15);
    }
#pragma unroll
    for (int i = 0; i < 4; ++i)
#pragma unroll
      for (int j = 0; j < 2; ++j)
        acc[i][j] = __builtin_amdgcn_wmma_f32_16x16x32_bf16(
            false, af[i], false, bfr[j], (short)0, acc[i][j], false, false);
    __syncthreads();   // all waves done with buf[cur] before TDM overwrites it
  }

  // epilogue: D layout lane -> col N (=l), vgpr v -> row M (= v + 8*gsel)
  float* ob = outp + (size_t)b * HCH * LSEQ;
#pragma unroll
  for (int i = 0; i < 4; ++i) {
#pragma unroll
    for (int j = 0; j < 2; ++j) {
      int l = l0 + wn * 32 + j * 16 + lr;
#pragma unroll
      for (int v = 0; v < 8; ++v) {
        int o = m0 + wm * 64 + i * 16 + gsel * 8 + v;
        ob[(size_t)o * LSEQ + l] = acc[i][j][v] + bias[o];
      }
    }
  }
}

// ---------------------------------------------------------------------------
extern "C" void kernel_launch(void* const* d_in, const int* in_sizes, int n_in,
                              void* d_out, int out_size, void* d_ws, size_t ws_size,
                              hipStream_t stream) {
  (void)in_sizes; (void)n_in; (void)out_size; (void)ws_size;
  const float* u = (const float*)d_in[0];   // (B,H,L)
  const float* k = (const float*)d_in[1];   // (1,H,L)
  const float* D = (const float*)d_in[2];   // (1,H)
  const float* W = (const float*)d_in[3];   // (H,H)
  const float* bias = (const float*)d_in[4];
  float* outp = (float*)d_out;

  char* p = (char*)d_ws;
  c32* tw = (c32*)p;                 p += (size_t)(NFFT / 2) * sizeof(c32);      // 64 KB
  c32* Kf = (c32*)p;                 p += (size_t)HCH * NFFT * sizeof(c32);      // 64 MB
  unsigned short* g  = (unsigned short*)p; p += (size_t)BATCH * HCH * LSEQ * 2;  // 64 MB
  unsigned short* gt = (unsigned short*)p; p += (size_t)BATCH * HCH * LSEQ * 2;  // 64 MB
  unsigned short* Wb = (unsigned short*)p; p += (size_t)HCH * HCH * 2;           // 0.5 MB

  s4_tw_init<<<(NFFT / 2 + 255) / 256, 256, 0, stream>>>(tw);
  s4_wprep<<<(HCH * HCH + 255) / 256, 256, 0, stream>>>(W, Wb);
  s4_kfft<<<HCH, FFT_T, NFFT * sizeof(c32), stream>>>(k, Kf, tw);
  s4_conv<<<BATCH * HCH, FFT_T, NFFT * sizeof(c32), stream>>>(u, Kf, D, g, tw);
  s4_transpose<<<dim3(LSEQ / 32, HCH / 32, BATCH), dim3(32, 8), 0, stream>>>(g, gt);
  s4_gemm<<<4 * (BATCH * LSEQ / BN), 256, 0, stream>>>((const __bf16*)Wb,
                                                       (const __bf16*)gt, bias, outp);
}
